// SelfAttention_75883482186271
// MI455X (gfx1250) — matrix operand order repaired and checked by reading, not compiled, not run on hardware
//
#include <hip/hip_runtime.h>
#include <math.h>

// ---------------------------------------------------------------------------
// Self-attention for MI455X (gfx1250): all GEMMs via v_wmma_f32_16x16x32_bf16.
// B=4, S=2048, D=1024. Compute-bound (~120 GFLOP vs ~80MB of HBM traffic),
// so everything runs through the WMMA pipe with bf16 operands / f32 accum.
// Per-batch working set fits in the 192MB L2 -> direct global fragment loads.
// ---------------------------------------------------------------------------

typedef __attribute__((ext_vector_type(16))) __bf16 v16bf;
typedef __attribute__((ext_vector_type(8)))  float  v8f;
typedef __attribute__((ext_vector_type(8)))  unsigned short v8us;
typedef unsigned short u16;

struct Frag32 {            // 32 bytes of bf16 operand data (16 elements/lane)
    v8us lo, hi;
};

__device__ inline u16 f2bf(float f) {   // round-to-nearest-even fp32 -> bf16
    unsigned u = __float_as_uint(f);
    u += 0x7fffu + ((u >> 16) & 1u);
    return (u16)(u >> 16);
}

// ----------------------------- converters ---------------------------------

__global__ __launch_bounds__(256) void cvt_f32_to_bf16(const float* __restrict__ in,
                                                       u16* __restrict__ out, int n) {
    int i = blockIdx.x * 256 + threadIdx.x;
    if (i < n) out[i] = f2bf(in[i]);
}

// W[d][e] fp32 (1024x1024) -> Wt[e][d] bf16 (n-major for WMMA B operand)
__global__ __launch_bounds__(256) void cvt_w_transpose(const float* __restrict__ W,
                                                       u16* __restrict__ Wt) {
    int i = blockIdx.x * 256 + threadIdx.x;   // 1M threads
    int e = i & 1023, d = i >> 10;
    Wt[(size_t)e * 1024 + d] = f2bf(W[(size_t)d * 1024 + e]);
}

// ----------------------------- WMMA GEMM -----------------------------------
// C[M,N] = scale * (A[M,K] @ B[K,N]) + bias, with B supplied n-major (Bt[N,K]).
// Block: 256 threads = 8 waves; block tile 128x128; wave tile 32x64 (2x4 frags).
// A-frag lane l: row = l&15 (+16*mf), two 16B chunks at k = kk+(l>>4)*8 and +16.
// B-frag lane l: col = l&15 (+16*nf), two 16B chunks at k = kk+(l>>4)*8 and +16.
__global__ __launch_bounds__(256)
void wmma_gemm_bf16(const u16* __restrict__ A, int lda,
                    const u16* __restrict__ Bt, int ldb, int K,
                    const float* __restrict__ bias, float scale,
                    float* __restrict__ outF, u16* __restrict__ outH,
                    int ldo, int vTrans) {
    const int lane  = threadIdx.x & 31;
    const int wave  = threadIdx.x >> 5;
    const int waveM = wave >> 1;              // 0..3
    const int waveN = wave & 1;               // 0..1
    const int mBase = blockIdx.y * 128 + waveM * 32;
    const int nBase = blockIdx.x * 128 + waveN * 64;
    const int lHalf = lane >> 4;              // 0|1
    const int lMod  = lane & 15;

    v8f acc[2][4];
#pragma unroll
    for (int mf = 0; mf < 2; ++mf)
#pragma unroll
        for (int nf = 0; nf < 4; ++nf)
#pragma unroll
            for (int i = 0; i < 8; ++i) acc[mf][nf][i] = 0.0f;

    const u16* aRow0 = A  + (size_t)(mBase + lMod) * lda;
    const u16* bCol0 = Bt + (size_t)(nBase + lMod) * ldb;

#pragma unroll 1
    for (int kk = 0; kk < K; kk += 32) {
        Frag32 a[2], b[4];
#pragma unroll
        for (int mf = 0; mf < 2; ++mf) {
            const u16* p = aRow0 + (size_t)mf * 16 * lda + kk + lHalf * 8;
            a[mf].lo = *(const v8us*)p;
            a[mf].hi = *(const v8us*)(p + 16);
        }
#pragma unroll
        for (int nf = 0; nf < 4; ++nf) {
            const u16* p = bCol0 + (size_t)nf * 16 * ldb + kk + lHalf * 8;
            b[nf].lo = *(const v8us*)p;
            b[nf].hi = *(const v8us*)(p + 16);
        }
        if (kk + 32 < K) {   // hint next K tile into cache (global_prefetch_b8)
            __builtin_prefetch(aRow0 + kk + 32, 0, 3);
            __builtin_prefetch(bCol0 + kk + 32, 0, 3);
        }
#pragma unroll
        for (int mf = 0; mf < 2; ++mf) {
            v16bf av = __builtin_bit_cast(v16bf, a[mf]);
#pragma unroll
            for (int nf = 0; nf < 4; ++nf) {
                v16bf bv = __builtin_bit_cast(v16bf, b[nf]);
                acc[mf][nf] = __builtin_amdgcn_wmma_f32_16x16x32_bf16(
                    false, av, false, bv, (short)0, acc[mf][nf], false, false);
            }
        }
    }

    // Epilogue. C/D layout: vgpr i -> row (l>>4)*8+i, col l&15.
#pragma unroll
    for (int mf = 0; mf < 2; ++mf)
#pragma unroll
        for (int nf = 0; nf < 4; ++nf) {
            int col = nBase + nf * 16 + lMod;
            float bsv = bias ? bias[col] : 0.0f;
#pragma unroll
            for (int i = 0; i < 8; ++i) {
                int row = mBase + mf * 16 + lHalf * 8 + i;
                float v = acc[mf][nf][i] * scale + bsv;
                if (outF) {
                    outF[(size_t)row * ldo + col] = v;
                } else if (vTrans) {
                    // V projection stored transposed per batch: Vt[b][e][s]
                    int bb = row >> 11, s = row & 2047;
                    outH[((size_t)bb * 1024 + col) * 2048 + s] = f2bf(v);
                } else {
                    outH[(size_t)row * ldo + col] = f2bf(v);
                }
            }
        }
}

// ----------------------------- row softmax ---------------------------------
// One block per row of a 2048-wide fp32 score matrix -> bf16 probabilities.
__global__ __launch_bounds__(256)
void softmax_rows(const float* __restrict__ S, u16* __restrict__ P, int n) {
    __shared__ float red[256];
    const int tid = threadIdx.x;
    const float* r = S + (size_t)blockIdx.x * n;

    float vals[8];
    float m = -INFINITY;
#pragma unroll
    for (int i = 0; i < 8; ++i) {
        vals[i] = r[tid + i * 256];
        m = fmaxf(m, vals[i]);
    }
    red[tid] = m; __syncthreads();
    for (int s = 128; s > 0; s >>= 1) {
        if (tid < s) red[tid] = fmaxf(red[tid], red[tid + s]);
        __syncthreads();
    }
    m = red[0]; __syncthreads();

    float l = 0.0f;
#pragma unroll
    for (int i = 0; i < 8; ++i) {
        vals[i] = __expf(vals[i] - m);
        l += vals[i];
    }
    red[tid] = l; __syncthreads();
    for (int s = 128; s > 0; s >>= 1) {
        if (tid < s) red[tid] += red[tid + s];
        __syncthreads();
    }
    float inv = 1.0f / red[0];

    u16* pr = P + (size_t)blockIdx.x * n;
#pragma unroll
    for (int i = 0; i < 8; ++i) pr[tid + i * 256] = f2bf(vals[i] * inv);
}

// ----------------------------- launcher ------------------------------------

extern "C" void kernel_launch(void* const* d_in, const int* in_sizes, int n_in,
                              void* d_out, int out_size, void* d_ws, size_t ws_size,
                              hipStream_t stream) {
    (void)in_sizes; (void)n_in; (void)out_size; (void)ws_size;

    const float* x  = (const float*)d_in[0];
    const float* Wq = (const float*)d_in[1];
    const float* bq = (const float*)d_in[2];
    const float* Wk = (const float*)d_in[3];
    const float* bk = (const float*)d_in[4];
    const float* Wv = (const float*)d_in[5];
    const float* bv = (const float*)d_in[6];
    float* out = (float*)d_out;

    const int M = 8192;                   // B*S
    const size_t NX = (size_t)M * 1024;   // elements in x / q / k / v

    // Scratch layout (~98.6 MB): x_bf | Wt_bf x3 | q_bf | k_bf | vt_bf |
    //                            scores_f32 (1 batch) | probs_bf16 (1 batch)
    char* ws = (char*)d_ws;
    u16*   x_bf  = (u16*)ws;
    u16*   wt_bf = (u16*)(ws + NX * 2);
    u16*   q_bf  = wt_bf + (size_t)3 * 1024 * 1024;
    u16*   k_bf  = q_bf + NX;
    u16*   vt_bf = k_bf + NX;
    float* sc    = (float*)(vt_bf + NX);
    u16*   attn  = (u16*)(sc + (size_t)2048 * 2048);

    dim3 blk(256);

    // 1) precision conversion (+ weight transpose for the WMMA B-operand)
    cvt_f32_to_bf16<<<(int)(NX / 256), blk, 0, stream>>>(x, x_bf, (int)NX);
    cvt_w_transpose<<<1024 * 1024 / 256, blk, 0, stream>>>(Wq, wt_bf);
    cvt_w_transpose<<<1024 * 1024 / 256, blk, 0, stream>>>(Wk, wt_bf + 1024 * 1024);
    cvt_w_transpose<<<1024 * 1024 / 256, blk, 0, stream>>>(Wv, wt_bf + 2 * 1024 * 1024);

    // 2) QKV projections: [8192,1024] @ [1024,1024] + bias (V stored transposed)
    dim3 gproj(1024 / 128, M / 128);
    wmma_gemm_bf16<<<gproj, blk, 0, stream>>>(x_bf, 1024, wt_bf,                1024, 1024,
                                              bq, 1.0f, nullptr, q_bf, 1024, 0);
    wmma_gemm_bf16<<<gproj, blk, 0, stream>>>(x_bf, 1024, wt_bf + 1024 * 1024,  1024, 1024,
                                              bk, 1.0f, nullptr, k_bf, 1024, 0);
    wmma_gemm_bf16<<<gproj, blk, 0, stream>>>(x_bf, 1024, wt_bf + 2*1024*1024,  1024, 1024,
                                              bv, 1.0f, nullptr, vt_bf, 0, 1);

    // 3) per-batch attention: scores (x 1/sqrt(1024)) -> softmax -> P @ V
    for (int b = 0; b < 4; ++b) {
        const size_t qo = (size_t)b * 2048 * 1024;       // q/k/out batch offset
        const size_t vo = (size_t)b * 1024 * 2048;       // vt batch offset
        wmma_gemm_bf16<<<dim3(2048 / 128, 2048 / 128), blk, 0, stream>>>(
            q_bf + qo, 1024, k_bf + qo, 1024, 1024,
            nullptr, 0.03125f, sc, nullptr, 2048, 0);
        softmax_rows<<<2048, blk, 0, stream>>>(sc, attn, 2048);
        wmma_gemm_bf16<<<dim3(1024 / 128, 2048 / 128), blk, 0, stream>>>(
            attn, 2048, vt_bf + vo, 2048, 2048,
            nullptr, 1.0f, out + qo, nullptr, 1024, 0);
    }
}
